// HeteroGraph_58179626992420
// MI455X (gfx1250) — matrix operand order repaired and checked by reading, not compiled, not run on hardware
//
#include <hip/hip_runtime.h>
#include <hip/hip_bf16.h>
#include <math.h>

#define NNODES 50000
#define NEDGES 800000
#define CH     128
#define CC     (CH * CH)

typedef __bf16 bf16_t;
typedef __attribute__((ext_vector_type(4)))  __bf16 v4bf;
typedef __attribute__((ext_vector_type(8)))  __bf16 v8bf;
typedef __attribute__((ext_vector_type(16))) __bf16 v16bf;
typedef __attribute__((ext_vector_type(8)))  float  v8f;

// ---------------------------------------------------------------- helpers
__device__ __forceinline__ __bf16 f2bf(float f) {
    // round-to-nearest-even fp32 -> bf16
    unsigned u = __float_as_uint(f);
    unsigned r = u + 0x7FFFu + ((u >> 16) & 1u);
    unsigned short h = (unsigned short)(r >> 16);
    return __builtin_bit_cast(__bf16, h);
}

__device__ __forceinline__ void atomicMaxF(float* addr, float val) {
    unsigned* ua = (unsigned*)addr;
    unsigned old = *ua;
    while (true) {
        float cur = __uint_as_float(old);
        if (!(val > cur)) break;
        unsigned prev = atomicCAS(ua, old, __float_as_uint(val));
        if (prev == old) break;
        old = prev;
    }
}

// Wave-level: acc[nt] += X[mbase:+16, 0:128] @ W[nt*16+n, :]^T
// X, W are bf16 row-major [*, 128]; pure b128 loads, zero conversions.
__device__ __forceinline__ void wave_gemm_acc(
    const bf16_t* __restrict__ X, const bf16_t* __restrict__ W,
    int mbase, int M, int lane, v8f acc[8])
{
    const int half = lane >> 4;            // 0 or 1
    int mrow = mbase + (lane & 15);
    if (mrow >= M) mrow = M - 1;           // clamp: garbage rows are never stored
    const v8bf* arow = (const v8bf*)(X + (size_t)mrow * CH);   // 16 chunks of 8 bf16

    // A layout: lane(half h) K-runs per kstep: [ks*32+8h, +8) and [ks*32+16+8h, +8)
    v16bf areg[4];
#pragma unroll
    for (int ks = 0; ks < 4; ++ks) {
        v8bf lo = arow[ks * 4 + half];
        v8bf hi = arow[ks * 4 + 2 + half];
        areg[ks] = __builtin_shufflevector(lo, hi,
                       0, 1, 2, 3, 4, 5, 6, 7, 8, 9, 10, 11, 12, 13, 14, 15);
    }
    const int ncol = lane & 15;
#pragma unroll
    for (int nt = 0; nt < 8; ++nt) {
        const v8bf* wrow = (const v8bf*)(W + (size_t)(nt * 16 + ncol) * CH);
        v8f c = acc[nt];
#pragma unroll
        for (int ks = 0; ks < 4; ++ks) {
            // B layout: lane(half h) K-runs: [ks*32+8h, +8) and [ks*32+16+8h, +8)
            v8bf lo = wrow[ks * 4 + half];
            v8bf hi = wrow[ks * 4 + 2 + half];
            v16bf breg = __builtin_shufflevector(lo, hi,
                             0, 1, 2, 3, 4, 5, 6, 7, 8, 9, 10, 11, 12, 13, 14, 15);
            c = __builtin_amdgcn_wmma_f32_16x16x32_bf16(
                    false, areg[ks], false, breg, (short)0, c, false, false);
        }
        acc[nt] = c;
    }
}

// ---------------------------------------------------------------- kernels
__global__ void fill_kernel(float* p, float v, long long n) {
    long long i = (long long)blockIdx.x * blockDim.x + threadIdx.x;
    if (i < n) p[i] = v;
}

// fp32 -> bf16 stream conversion (4 elems/thread)
__global__ void cvt_bf16_kernel(const float4* __restrict__ src,
                                v4bf* __restrict__ dst, long long n4) {
    long long i = (long long)blockIdx.x * blockDim.x + threadIdx.x;
    if (i >= n4) return;
    float4 v = src[i];
    v4bf o;
    o[0] = f2bf(v.x); o[1] = f2bf(v.y); o[2] = f2bf(v.z); o[3] = f2bf(v.w);
    dst[i] = o;
}

// agg[dst] += x_src[src]  (one wave per edge, 4 floats per lane)
__global__ void sage_scatter_kernel(const float* __restrict__ xsrc,
                                    const int* __restrict__ ei, int E,
                                    float* __restrict__ agg) {
    long long t = (long long)blockIdx.x * blockDim.x + threadIdx.x;
    int e = (int)(t >> 5);
    int lane = (int)(t & 31);
    if (e >= E) return;
    int src = ei[e];
    int dst = ei[E + e];
    const float4* xs = (const float4*)(xsrc + (size_t)src * CH);
    float4 v = xs[lane];
    float* ap = agg + (size_t)dst * CH + lane * 4;
    atomicAdd(ap + 0, v.x); atomicAdd(ap + 1, v.y);
    atomicAdd(ap + 2, v.z); atomicAdd(ap + 3, v.w);
}

// out_bf = bf16( relu(LN(A1@W1.T + bias + A2@W2.T)) ) ; fully fused SAGE layer
__global__ __launch_bounds__(256) void sage_fused_kernel(
    const bf16_t* __restrict__ A1, const bf16_t* __restrict__ W1,
    const bf16_t* __restrict__ A2, const bf16_t* __restrict__ W2,
    const float* __restrict__ bias,
    const float* __restrict__ gamma, const float* __restrict__ beta,
    bf16_t* __restrict__ out_bf, int M)
{
    const int lane  = threadIdx.x & 31;
    const int wave  = threadIdx.x >> 5;
    const int mbase = blockIdx.x * 128 + wave * 16;
    if (mbase >= M) return;                 // wave-uniform: EXEC stays all-ones
    const int col  = lane & 15;
    const int half = lane >> 4;

    v8f acc[8];
#pragma unroll
    for (int nt = 0; nt < 8; ++nt) {
        float bv = bias[nt * 16 + col];
#pragma unroll
        for (int r = 0; r < 8; ++r) acc[nt][r] = bv;
    }
    wave_gemm_acc(A1, W1, mbase, M, lane, acc);
    wave_gemm_acc(A2, W2, mbase, M, lane, acc);

#pragma unroll
    for (int r = 0; r < 8; ++r) {
        float s = 0.f, s2 = 0.f;
#pragma unroll
        for (int nt = 0; nt < 8; ++nt) { float v = acc[nt][r]; s += v; s2 += v * v; }
#pragma unroll
        for (int off = 1; off < 16; off <<= 1) {   // reduce within 16-lane halves
            s  += __shfl_xor(s,  off, 32);
            s2 += __shfl_xor(s2, off, 32);
        }
        float mean = s * (1.0f / CH);
        float var  = s2 * (1.0f / CH) - mean * mean;
        float inv  = rsqrtf(var + 1e-5f);
        int row = mbase + r + 8 * half;
        if (row < M) {
#pragma unroll
            for (int nt = 0; nt < 8; ++nt) {
                int c = nt * 16 + col;
                float v = (acc[nt][r] - mean) * inv * gamma[c] + beta[c];
                out_bf[(size_t)row * CH + c] = f2bf(fmaxf(v, 0.0f));
            }
        }
    }
}

// h = X@W.T (optional fp32 store), alpha = h @ avec  per row
__global__ __launch_bounds__(256) void gat_transform_kernel(
    const bf16_t* __restrict__ X, const bf16_t* __restrict__ W,
    const float* __restrict__ avec,
    float* __restrict__ h_out, float* __restrict__ alpha_out, int M)
{
    const int lane  = threadIdx.x & 31;
    const int wave  = threadIdx.x >> 5;
    const int mbase = blockIdx.x * 128 + wave * 16;
    if (mbase >= M) return;
    const int col  = lane & 15;
    const int half = lane >> 4;

    v8f acc[8];
#pragma unroll
    for (int nt = 0; nt < 8; ++nt)
#pragma unroll
        for (int r = 0; r < 8; ++r) acc[nt][r] = 0.0f;
    wave_gemm_acc(X, W, mbase, M, lane, acc);

#pragma unroll
    for (int r = 0; r < 8; ++r) {
        float dot = 0.f;
#pragma unroll
        for (int nt = 0; nt < 8; ++nt) dot += acc[nt][r] * avec[nt * 16 + col];
#pragma unroll
        for (int off = 1; off < 16; off <<= 1) dot += __shfl_xor(dot, off, 32);
        int row = mbase + r + 8 * half;
        if (row < M) {
            if (h_out) {
#pragma unroll
                for (int nt = 0; nt < 8; ++nt)
                    h_out[(size_t)row * CH + nt * 16 + col] = acc[nt][r];
            }
            if (col == 0) alpha_out[row] = dot;
        }
    }
}

// pass A: e = leaky_relu(as[src]+ad[dst]); store; atomic max per dst
__global__ void gat_edge_max_kernel(const float* __restrict__ as,
                                    const float* __restrict__ ad,
                                    const int* __restrict__ ei, int E,
                                    float* __restrict__ ev, float* __restrict__ m) {
    int e = blockIdx.x * blockDim.x + threadIdx.x;
    if (e >= E) return;
    int s = ei[e], d = ei[E + e];
    float x  = as[s] + ad[d];
    float lr = x > 0.0f ? x : 0.2f * x;
    ev[e] = lr;
    atomicMaxF(m + d, lr);
}

// pass B: ex = exp(e - m[dst]); store; atomic sum per dst
__global__ void gat_edge_exp_kernel(const float* __restrict__ m,
                                    const int* __restrict__ ei, int E,
                                    float* __restrict__ ev, float* __restrict__ denom) {
    int e = blockIdx.x * blockDim.x + threadIdx.x;
    if (e >= E) return;
    int d = ei[E + e];
    float ex = __expf(ev[e] - m[d]);
    ev[e] = ex;
    atomicAdd(denom + d, ex);
}

// pass C: out[dst] += (ex/denom[dst]) * hs[src]
__global__ void gat_scatter_kernel(const float* __restrict__ hs,
                                   const float* __restrict__ ev,
                                   const float* __restrict__ denom,
                                   const int* __restrict__ ei, int E,
                                   float* __restrict__ out) {
    long long t = (long long)blockIdx.x * blockDim.x + threadIdx.x;
    int e = (int)(t >> 5);
    int lane = (int)(t & 31);
    if (e >= E) return;
    int s = ei[e], d = ei[E + e];
    float alpha = ev[e] / (denom[d] + 1e-16f);
    const float4* hp = (const float4*)(hs + (size_t)s * CH);
    float4 v = hp[lane];
    float* op = out + (size_t)d * CH + lane * 4;
    atomicAdd(op + 0, alpha * v.x); atomicAdd(op + 1, alpha * v.y);
    atomicAdd(op + 2, alpha * v.z); atomicAdd(op + 3, alpha * v.w);
}

// out = relu(LN(raw + b)) ; one wave per row, fp32 output
__global__ __launch_bounds__(256) void bias_ln_relu_kernel(
    const float* __restrict__ raw, const float* __restrict__ b,
    const float* __restrict__ gamma, const float* __restrict__ beta,
    float* __restrict__ out, int M)
{
    int lane = threadIdx.x & 31;
    int wave = threadIdx.x >> 5;
    int row  = blockIdx.x * 8 + wave;
    if (row >= M) return;
    float4 v  = ((const float4*)(raw + (size_t)row * CH))[lane];
    float4 bb = ((const float4*)b)[lane];
    float x0 = v.x + bb.x, x1 = v.y + bb.y, x2 = v.z + bb.z, x3 = v.w + bb.w;
    float s  = x0 + x1 + x2 + x3;
    float s2 = x0 * x0 + x1 * x1 + x2 * x2 + x3 * x3;
#pragma unroll
    for (int off = 1; off < 32; off <<= 1) {
        s  += __shfl_xor(s,  off, 32);
        s2 += __shfl_xor(s2, off, 32);
    }
    float mean = s * (1.0f / CH);
    float var  = s2 * (1.0f / CH) - mean * mean;
    float inv  = rsqrtf(var + 1e-5f);
    float4 g4 = ((const float4*)gamma)[lane];
    float4 b4 = ((const float4*)beta)[lane];
    float* op = out + (size_t)row * CH + lane * 4;
    op[0] = fmaxf((x0 - mean) * inv * g4.x + b4.x, 0.0f);
    op[1] = fmaxf((x1 - mean) * inv * g4.y + b4.y, 0.0f);
    op[2] = fmaxf((x2 - mean) * inv * g4.z + b4.z, 0.0f);
    op[3] = fmaxf((x3 - mean) * inv * g4.w + b4.w, 0.0f);
}

// ---------------------------------------------------------------- launch
extern "C" void kernel_launch(void* const* d_in, const int* in_sizes, int n_in,
                              void* d_out, int out_size, void* d_ws, size_t ws_size,
                              hipStream_t stream) {
    const float* x_paper  = (const float*)d_in[0];
    const float* x_author = (const float*)d_in[1];
    const int*   ei_w     = (const int*)d_in[2];
    const int*   ei_b     = (const int*)d_in[3];
    const float* sw_Wl = (const float*)d_in[4];
    const float* sw_bl = (const float*)d_in[5];
    const float* sw_Wr = (const float*)d_in[6];
    const float* sb_Wl = (const float*)d_in[7];
    const float* sb_bl = (const float*)d_in[8];
    const float* sb_Wr = (const float*)d_in[9];
    const float* gw_Ws = (const float*)d_in[10];
    const float* gw_Wd = (const float*)d_in[11];
    const float* gw_as = (const float*)d_in[12];
    const float* gw_ad = (const float*)d_in[13];
    const float* gw_b  = (const float*)d_in[14];
    const float* gb_Ws = (const float*)d_in[15];
    const float* gb_Wd = (const float*)d_in[16];
    const float* gb_as = (const float*)d_in[17];
    const float* gb_ad = (const float*)d_in[18];
    const float* gb_b  = (const float*)d_in[19];
    const float* ln0p_g = (const float*)d_in[20];
    const float* ln0p_b = (const float*)d_in[21];
    const float* ln0a_g = (const float*)d_in[22];
    const float* ln0a_b = (const float*)d_in[23];
    const float* ln1p_g = (const float*)d_in[24];
    const float* ln1p_b = (const float*)d_in[25];
    const float* ln1a_g = (const float*)d_in[26];
    const float* ln1a_b = (const float*)d_in[27];

    const size_t NC = (size_t)NNODES * CH;

    // ---- workspace carve-up (fp32 region, then bf16 region) ----
    float* wsf = (float*)d_ws;
    float* agg_p = wsf;                 // later reused as p2raw
    float* agg_a = wsf + NC;            // later reused as a2raw
    float* hs_w  = wsf + 2 * NC;
    float* hs_b  = wsf + 3 * NC;
    float* sc    = wsf + 4 * NC;
    float* as_w = sc + 0 * NNODES;
    float* ad_w = sc + 1 * NNODES;
    float* as_b = sc + 2 * NNODES;
    float* ad_b = sc + 3 * NNODES;
    float* m_w  = sc + 4 * NNODES;      // m_w, m_b contiguous
    float* m_b  = sc + 5 * NNODES;
    float* dn_w = sc + 6 * NNODES;      // dn_w, dn_b contiguous
    float* dn_b = sc + 7 * NNODES;
    float* ev_w = sc + 8 * NNODES;
    float* ev_b = ev_w + NEDGES;

    bf16_t* wsb = (bf16_t*)(ev_b + NEDGES);
    bf16_t* xp_bf   = wsb;
    bf16_t* xa_bf   = wsb + NC;
    bf16_t* aggp_bf = wsb + 2 * NC;
    bf16_t* agga_bf = wsb + 3 * NC;
    bf16_t* p_bf    = wsb + 4 * NC;
    bf16_t* a_bf    = wsb + 5 * NC;
    bf16_t* wmats   = wsb + 6 * NC;     // 8 weight matrices, bf16
    bf16_t* swWl_bf = wmats + 0 * CC;
    bf16_t* swWr_bf = wmats + 1 * CC;
    bf16_t* sbWl_bf = wmats + 2 * CC;
    bf16_t* sbWr_bf = wmats + 3 * CC;
    bf16_t* gwWs_bf = wmats + 4 * CC;
    bf16_t* gwWd_bf = wmats + 5 * CC;
    bf16_t* gbWs_bf = wmats + 6 * CC;
    bf16_t* gbWd_bf = wmats + 7 * CC;

    dim3 b256(256);
    auto nb = [](long long n) { return dim3((unsigned)((n + 255) / 256)); };
    const long long scatterThreads = (long long)NEDGES * 32;
    dim3 gemmGrid((NNODES + 127) / 128);
    const long long NC4 = (long long)(NC / 4);
    const long long CC4 = CC / 4;

    // ---- one-shot bf16 conversions (weights + input features) ----
    cvt_bf16_kernel<<<nb(NC4), b256, 0, stream>>>((const float4*)x_paper,  (v4bf*)xp_bf, NC4);
    cvt_bf16_kernel<<<nb(NC4), b256, 0, stream>>>((const float4*)x_author, (v4bf*)xa_bf, NC4);
    const float*  wsrc[8] = {sw_Wl, sw_Wr, sb_Wl, sb_Wr, gw_Ws, gw_Wd, gb_Ws, gb_Wd};
    bf16_t*       wdst[8] = {swWl_bf, swWr_bf, sbWl_bf, sbWr_bf, gwWs_bf, gwWd_bf, gbWs_bf, gbWd_bf};
    for (int i = 0; i < 8; ++i)
        cvt_bf16_kernel<<<nb(CC4), b256, 0, stream>>>((const float4*)wsrc[i], (v4bf*)wdst[i], CC4);

    // ---- layer 0: SAGE ----
    fill_kernel<<<nb(2 * (long long)NC), b256, 0, stream>>>(agg_p, 0.0f, 2 * (long long)NC);
    sage_scatter_kernel<<<nb(scatterThreads), b256, 0, stream>>>(x_author, ei_w, NEDGES, agg_p);
    sage_scatter_kernel<<<nb(scatterThreads), b256, 0, stream>>>(x_paper,  ei_b, NEDGES, agg_a);
    cvt_bf16_kernel<<<nb(NC4), b256, 0, stream>>>((const float4*)agg_p, (v4bf*)aggp_bf, NC4);
    cvt_bf16_kernel<<<nb(NC4), b256, 0, stream>>>((const float4*)agg_a, (v4bf*)agga_bf, NC4);

    sage_fused_kernel<<<gemmGrid, b256, 0, stream>>>(aggp_bf, swWl_bf, xp_bf, swWr_bf, sw_bl,
                                                     ln0p_g, ln0p_b, p_bf, NNODES);
    sage_fused_kernel<<<gemmGrid, b256, 0, stream>>>(agga_bf, sbWl_bf, xa_bf, sbWr_bf, sb_bl,
                                                     ln0a_g, ln0a_b, a_bf, NNODES);

    // ---- layer 1: GAT ----
    fill_kernel<<<nb(2 * (long long)NC), b256, 0, stream>>>(agg_p, 0.0f, 2 * (long long)NC); // p2raw/a2raw
    fill_kernel<<<nb(2 * NNODES), b256, 0, stream>>>(dn_w, 0.0f, 2 * NNODES);
    fill_kernel<<<nb(2 * NNODES), b256, 0, stream>>>(m_w, -INFINITY, 2 * NNODES);

    gat_transform_kernel<<<gemmGrid, b256, 0, stream>>>(a_bf, gwWs_bf, gw_as, hs_w,   as_w, NNODES);
    gat_transform_kernel<<<gemmGrid, b256, 0, stream>>>(p_bf, gwWd_bf, gw_ad, nullptr, ad_w, NNODES);
    gat_transform_kernel<<<gemmGrid, b256, 0, stream>>>(p_bf, gbWs_bf, gb_as, hs_b,   as_b, NNODES);
    gat_transform_kernel<<<gemmGrid, b256, 0, stream>>>(a_bf, gbWd_bf, gb_ad, nullptr, ad_b, NNODES);

    gat_edge_max_kernel<<<nb(NEDGES), b256, 0, stream>>>(as_w, ad_w, ei_w, NEDGES, ev_w, m_w);
    gat_edge_max_kernel<<<nb(NEDGES), b256, 0, stream>>>(as_b, ad_b, ei_b, NEDGES, ev_b, m_b);
    gat_edge_exp_kernel<<<nb(NEDGES), b256, 0, stream>>>(m_w, ei_w, NEDGES, ev_w, dn_w);
    gat_edge_exp_kernel<<<nb(NEDGES), b256, 0, stream>>>(m_b, ei_b, NEDGES, ev_b, dn_b);
    gat_scatter_kernel<<<nb(scatterThreads), b256, 0, stream>>>(hs_w, ev_w, dn_w, ei_w, NEDGES, agg_p);
    gat_scatter_kernel<<<nb(scatterThreads), b256, 0, stream>>>(hs_b, ev_b, dn_b, ei_b, NEDGES, agg_a);

    float* outp = (float*)d_out;
    dim3 lnGrid((NNODES + 7) / 8);
    bias_ln_relu_kernel<<<lnGrid, b256, 0, stream>>>(agg_p, gw_b, ln1p_g, ln1p_b, outp,      NNODES);
    bias_ln_relu_kernel<<<lnGrid, b256, 0, stream>>>(agg_a, gb_b, ln1a_g, ln1a_b, outp + NC, NNODES);
}
